// MultiDimVQVAE_17738214933195
// MI455X (gfx1250) — matrix operand order repaired and checked by reading, not compiled, not run on hardware
//
#include <hip/hip_runtime.h>
#include <hip/hip_bf16.h>
#include <math.h>

// ---------------------------------------------------------------------------
// MultiDimVQVAE forward, CDNA5 (gfx1250, wave32) WMMA bf16 implementation.
//
//   1. f32 -> bf16 conversion of x, enc_w, dec_w, codebooks
//   2. e_sq[s,k] = sum_d codebook^2            (wave reduction)
//   3. z = x @ enc_w^T + enc_b                  (WMMA bf16, ping-pong pipeline)
//   4. codes = argmin_k (e_sq[k] - 2*z.cb^T)    (WMMA bf16, fused argmin,
//                                                ping-pong B tiles)
//   5. gather quantized (f32 out + bf16 copy), codes as float out
//   6. x_recon = q @ dec_w^T + dec_b            (WMMA bf16, f32 out)
//   7. histogram + perplexity
// ---------------------------------------------------------------------------

typedef __attribute__((ext_vector_type(16))) __bf16         v16bf;
typedef __attribute__((ext_vector_type(8)))  __bf16         v8bf;
typedef __attribute__((ext_vector_type(8)))  float          v8f;
typedef __attribute__((ext_vector_type(8)))  unsigned short v8u;
typedef __attribute__((ext_vector_type(4)))  unsigned short v4u;

#define Bsz 4096
#define INDIM 2048
#define KCODE 8192
#define DDIM 256
#define SSPL 8
#define SD 2048   // S * D

// Scheduling directives (IGroupLP): mask 0x20 = VMEM read, 0x8 = MFMA/WMMA.
#if defined(__has_builtin)
#if __has_builtin(__builtin_amdgcn_sched_group_barrier)
#define SCHED_GB(m, s, id) __builtin_amdgcn_sched_group_barrier((m), (s), (id))
#else
#define SCHED_GB(m, s, id)
#endif
#else
#define SCHED_GB(m, s, id)
#endif

__device__ __forceinline__ unsigned short f32_to_bf16(float f) {
  unsigned u = __float_as_uint(f);
  unsigned r = u + 0x7FFFu + ((u >> 16) & 1u);   // round-to-nearest-even
  return (unsigned short)(r >> 16);
}

// A-fragment (16x32 bf16, MxK). `base` points at element (m0, k0) of a
// row-major matrix with leading dim `ld` (in elements).
// Lane L (m = L&15, h = L>>4): K = h*16..h*16+15.
__device__ __forceinline__ v16bf load_a16(const unsigned short* base, int ld, int lane) {
  int m = lane & 15, h = lane >> 4;
  const unsigned short* p = base + (size_t)m * ld + h * 16;
  v8bf lo = *reinterpret_cast<const v8bf*>(p);
  v8bf hi = *reinterpret_cast<const v8bf*>(p + 8);
  v16bf r;
#pragma unroll
  for (int i = 0; i < 8; ++i) { r[i] = lo[i]; r[i + 8] = hi[i]; }
  return r;
}

// B-fragment (32x16 bf16, KxN) for an NT GEMM: B^T is row-major [N x K].
// `base` points at element (n0, k0) of B^T. Lane L (n = L&15, h = L>>4)
// holds B^T[n][k0 + h*16 .. +15]  -> one contiguous 32-byte load.
__device__ __forceinline__ v16bf load_b16(const unsigned short* base, int ld, int lane) {
  const unsigned short* p = base + (size_t)(lane & 15) * ld + (lane >> 4) * 16;
  return *reinterpret_cast<const v16bf*>(p);
}

// ---------------------------------------------------------------------------
// f32 -> bf16 conversion (vectorized x4)
// ---------------------------------------------------------------------------
__global__ void cvt_f32_bf16_kernel(const float* __restrict__ in,
                                    unsigned short* __restrict__ out, int n4) {
  int i = blockIdx.x * blockDim.x + threadIdx.x;
  if (i >= n4) return;
  float4 v = reinterpret_cast<const float4*>(in)[i];
  v4u o;
  o[0] = f32_to_bf16(v.x); o[1] = f32_to_bf16(v.y);
  o[2] = f32_to_bf16(v.z); o[3] = f32_to_bf16(v.w);
  reinterpret_cast<v4u*>(out)[i] = o;
}

// ---------------------------------------------------------------------------
// e_sq[s,k] = sum_d codebooks[s,k,d]^2  (one wave per row, 65536 rows)
// ---------------------------------------------------------------------------
__global__ void esq_kernel(const float* __restrict__ cb, float* __restrict__ esq) {
  int rid  = blockIdx.x * 8 + (threadIdx.x >> 5);
  int lane = threadIdx.x & 31;
  const float4* p = reinterpret_cast<const float4*>(cb + (size_t)rid * DDIM);
  float4 a = p[lane * 2], c = p[lane * 2 + 1];
  float sum = a.x * a.x + a.y * a.y + a.z * a.z + a.w * a.w +
              c.x * c.x + c.y * c.y + c.z * c.z + c.w * c.w;
#pragma unroll
  for (int m = 16; m >= 1; m >>= 1) sum += __shfl_xor(sum, m, 32);
  if (lane == 0) esq[rid] = sum;
}

// ---------------------------------------------------------------------------
// NT GEMM:  C[M,N] = A[M,K] (bf16 row-major) * Bw[N,K]^T (bf16 row-major) + bias
// Workgroup = 8 waves -> 128(M) x 64(N) tile; wave = 16(M) x 64(N).
// Copy-free ping-pong software pipeline over K (two buffer sets P/Q, manual
// unroll by 2): buffer loads are issued a half-iteration (4 WMMAs) before use.
// out_bf16 != 0 -> store bf16 (ushort), else store f32.
// ---------------------------------------------------------------------------
__global__ __attribute__((amdgpu_waves_per_eu(2)))
void gemm_nt_bias_kernel(const unsigned short* __restrict__ A,
                         const unsigned short* __restrict__ Bw,
                         const float* __restrict__ bias,
                         void* __restrict__ Cout,
                         int K, int ldc, int out_bf16) {
  int lane = threadIdx.x & 31;
  int m0 = blockIdx.y * 128 + (threadIdx.x >> 5) * 16;
  int n0 = blockIdx.x * 64;

  const unsigned short* Abase = A + (size_t)m0 * K;
  const unsigned short* Bbase = Bw + (size_t)n0 * K;

  v8f acc[4];
#pragma unroll
  for (int t = 0; t < 4; ++t)
#pragma unroll
    for (int i = 0; i < 8; ++i) acc[t][i] = 0.0f;

  // prologue: P buffer <- k = 0
  v16bf aP = load_a16(Abase, K, lane);
  v16bf bP[4];
#pragma unroll
  for (int t = 0; t < 4; ++t)
    bP[t] = load_b16(Bbase + (size_t)(t * 16) * K, K, lane);

#pragma unroll 1
  for (int k0 = 0; k0 < K; k0 += 64) {   // K is a multiple of 64
    int k1 = k0 + 32;
    int k2 = (k0 + 64 < K) ? (k0 + 64) : 0;   // last iter: harmless reload

    // load Q <- k1
    v16bf aQ = load_a16(Abase + k1, K, lane);
    v16bf bQ[4];
#pragma unroll
    for (int t = 0; t < 4; ++t)
      bQ[t] = load_b16(Bbase + (size_t)(t * 16) * K + k1, K, lane);

    // compute P (k0)
#pragma unroll
    for (int t = 0; t < 4; ++t)
      acc[t] = __builtin_amdgcn_wmma_f32_16x16x32_bf16(
          false, aP, false, bP[t], (short)0, acc[t], false, false);

    // load P <- k2
    aP = load_a16(Abase + k2, K, lane);
#pragma unroll
    for (int t = 0; t < 4; ++t)
      bP[t] = load_b16(Bbase + (size_t)(t * 16) * K + k2, K, lane);

    // compute Q (k1)
#pragma unroll
    for (int t = 0; t < 4; ++t)
      acc[t] = __builtin_amdgcn_wmma_f32_16x16x32_bf16(
          false, aQ, false, bQ[t], (short)0, acc[t], false, false);

    // enforce 2-load : 1-wmma interleave (20 VMEM reads, 8 WMMAs / iter)
#pragma unroll
    for (int g = 0; g < 8; ++g) {
      SCHED_GB(0x20, 2, 0);
      SCHED_GB(0x08, 1, 0);
    }
    SCHED_GB(0x20, 4, 0);
  }

  int col = lane & 15, h = lane >> 4;
#pragma unroll
  for (int t = 0; t < 4; ++t) {
    int n = n0 + t * 16 + col;
    float bv = bias[n];
#pragma unroll
    for (int r = 0; r < 8; ++r) {
      int row = m0 + h * 8 + r;
      float v = acc[t][r] + bv;
      if (out_bf16)
        ((unsigned short*)Cout)[(size_t)row * ldc + n] = f32_to_bf16(v);
      else
        ((float*)Cout)[(size_t)row * ldc + n] = v;
    }
  }
}

// ---------------------------------------------------------------------------
// Fused distance + argmin:
//   codes[b,s] = argmin_k ( e_sq[s,k] - 2 * <z[b,s,:], cb[s,k,:]> )
// One wave owns 16 rows of one split; A-fragments (16 x 256 bf16 = 64 VGPRs)
// stay resident while scanning all 8192 codes in 16-wide N tiles.
// Copy-free ping-pong over code tiles (two B buffer sets, manual unroll by 2):
// a tile's 16 x b128 loads issue a half-iteration (8 WMMAs) before use.
// 2048 waves total (8 splits x 256 row-tiles).
// ---------------------------------------------------------------------------
__global__ __attribute__((amdgpu_waves_per_eu(1)))
void vq_dist_argmin_kernel(const unsigned short* __restrict__ Zbf,
                           const float* __restrict__ esq,
                           const unsigned short* __restrict__ CBbf,
                           int* __restrict__ codes) {
  int wid  = blockIdx.x * 8 + (threadIdx.x >> 5);
  int lane = threadIdx.x & 31;
  int s  = wid >> 8;           // 0..7 (all waves of a block share s)
  int m0 = (wid & 255) * 16;   // row tile

  const unsigned short* Abase = Zbf + (size_t)m0 * SD + s * DDIM;
  v16bf afr[8];
#pragma unroll
  for (int kk = 0; kk < 8; ++kk) afr[kk] = load_a16(Abase + kk * 32, SD, lane);

  const unsigned short* CB = CBbf + (size_t)s * KCODE * DDIM;
  const float* es = esq + (size_t)s * KCODE;

  float minv[8];
  int   mini[8];
#pragma unroll
  for (int r = 0; r < 8; ++r) { minv[r] = 3.4e38f; mini[r] = 0; }

  int col = lane & 15;

  // prologue: P buffer <- tile n = 0
  v16bf bP[8];
#pragma unroll
  for (int kk = 0; kk < 8; ++kk)
    bP[kk] = load_b16(CB + kk * 32, DDIM, lane);

#pragma unroll 1
  for (int n0 = 0; n0 < KCODE; n0 += 32) {
    int n1 = n0 + 16;
    int n2 = (n0 + 32 < KCODE) ? (n0 + 32) : 0;   // last iter: harmless reload

    // load Q <- tile n1
    v16bf bQ[8];
#pragma unroll
    for (int kk = 0; kk < 8; ++kk)
      bQ[kk] = load_b16(CB + (size_t)n1 * DDIM + kk * 32, DDIM, lane);

    // compute P (tile n0)
    v8f accP;
#pragma unroll
    for (int i = 0; i < 8; ++i) accP[i] = 0.0f;
#pragma unroll
    for (int kk = 0; kk < 8; ++kk)
      accP = __builtin_amdgcn_wmma_f32_16x16x32_bf16(
          false, afr[kk], false, bP[kk], (short)0, accP, false, false);
    {
      float e  = es[n0 + col];
      int  idx = n0 + col;
#pragma unroll
      for (int r = 0; r < 8; ++r) {
        float sc = e - 2.0f * accP[r];
        if (sc < minv[r] || (sc == minv[r] && idx < mini[r])) {
          minv[r] = sc; mini[r] = idx;
        }
      }
    }

    // load P <- tile n2
#pragma unroll
    for (int kk = 0; kk < 8; ++kk)
      bP[kk] = load_b16(CB + (size_t)n2 * DDIM + kk * 32, DDIM, lane);

    // compute Q (tile n1)
    v8f accQ;
#pragma unroll
    for (int i = 0; i < 8; ++i) accQ[i] = 0.0f;
#pragma unroll
    for (int kk = 0; kk < 8; ++kk)
      accQ = __builtin_amdgcn_wmma_f32_16x16x32_bf16(
          false, afr[kk], false, bQ[kk], (short)0, accQ, false, false);
    {
      float e  = es[n1 + col];
      int  idx = n1 + col;
#pragma unroll
      for (int r = 0; r < 8; ++r) {
        float sc = e - 2.0f * accQ[r];
        if (sc < minv[r] || (sc == minv[r] && idx < mini[r])) {
          minv[r] = sc; mini[r] = idx;
        }
      }
    }

    // enforce 2-load : 1-wmma interleave (34 VMEM reads, 16 WMMAs / iter)
#pragma unroll
    for (int g = 0; g < 16; ++g) {
      SCHED_GB(0x20, 2, 0);
      SCHED_GB(0x08, 1, 0);
    }
    SCHED_GB(0x20, 2, 0);
  }

  // argmin across the 16 lanes of each half-wave (xor masks stay in-group)
#pragma unroll
  for (int r = 0; r < 8; ++r) {
    float v = minv[r];
    int   i = mini[r];
#pragma unroll
    for (int m = 8; m >= 1; m >>= 1) {
      float ov = __shfl_xor(v, m, 32);
      int   oi = __shfl_xor(i, m, 32);
      if (ov < v || (ov == v && oi < i)) { v = ov; i = oi; }
    }
    if (col == 0) {
      int row = m0 + (lane >> 4) * 8 + r;
      codes[row * SSPL + s] = i;
    }
  }
}

// ---------------------------------------------------------------------------
// Gather: quantized[b, s*256: ] = codebooks[s, codes[b,s], :]
// Writes f32 (output), bf16 (decoder input), and codes as float (output).
// One wave per (b,s) row; 32768 rows.
// ---------------------------------------------------------------------------
__global__ void vq_gather_kernel(const int* __restrict__ codes,
                                 const float* __restrict__ codebooks,
                                 float* __restrict__ quant_out,
                                 unsigned short* __restrict__ qbf,
                                 float* __restrict__ codes_out) {
  int rid  = blockIdx.x * 8 + (threadIdx.x >> 5);
  int lane = threadIdx.x & 31;
  int b = rid >> 3, s = rid & 7;
  int code = codes[rid];
  const float4* src = reinterpret_cast<const float4*>(
      codebooks + ((size_t)s * KCODE + code) * DDIM);
  size_t dstoff = (size_t)b * SD + s * DDIM;
  float4* dq = reinterpret_cast<float4*>(quant_out + dstoff);
  float4 v0 = src[lane * 2], v1 = src[lane * 2 + 1];
  dq[lane * 2] = v0; dq[lane * 2 + 1] = v1;
  v8u pk;
  pk[0] = f32_to_bf16(v0.x); pk[1] = f32_to_bf16(v0.y);
  pk[2] = f32_to_bf16(v0.z); pk[3] = f32_to_bf16(v0.w);
  pk[4] = f32_to_bf16(v1.x); pk[5] = f32_to_bf16(v1.y);
  pk[6] = f32_to_bf16(v1.z); pk[7] = f32_to_bf16(v1.w);
  *reinterpret_cast<v8u*>(qbf + dstoff + lane * 8) = pk;
  if (lane == 0) codes_out[rid] = (float)code;
}

// ---------------------------------------------------------------------------
// Histogram + perplexity
// ---------------------------------------------------------------------------
__global__ void zero_i32_kernel(int* __restrict__ p, int n) {
  int i = blockIdx.x * blockDim.x + threadIdx.x;
  if (i < n) p[i] = 0;
}

__global__ void hist_kernel(const int* __restrict__ codes, int* __restrict__ counts) {
  int i = blockIdx.x * blockDim.x + threadIdx.x;
  if (i < Bsz * SSPL) {
    int s = i & 7;
    atomicAdd(&counts[s * KCODE + codes[i]], 1);
  }
}

__global__ void ppl_kernel(const int* __restrict__ counts, float* __restrict__ out) {
  __shared__ float red[256];
  int tid = threadIdx.x;
  float acc = 0.0f;
  for (int s = 0; s < SSPL; ++s) {
    float part = 0.0f;
    for (int k = tid; k < KCODE; k += 256) {
      float p = (float)counts[s * KCODE + k] * (1.0f / (float)Bsz);
      part += p * logf(p + 1e-10f);
    }
    red[tid] = part;
    __syncthreads();
    for (int off = 128; off > 0; off >>= 1) {
      if (tid < off) red[tid] += red[tid + off];
      __syncthreads();
    }
    if (tid == 0) acc += expf(-red[0]);
    __syncthreads();
  }
  if (tid == 0) *out = acc * (1.0f / (float)SSPL);
}

// ---------------------------------------------------------------------------
// Launch
// ---------------------------------------------------------------------------
extern "C" void kernel_launch(void* const* d_in, const int* in_sizes, int n_in,
                              void* d_out, int out_size, void* d_ws, size_t ws_size,
                              hipStream_t stream) {
  (void)in_sizes; (void)n_in; (void)out_size; (void)ws_size;

  const float* x    = (const float*)d_in[0];   // [4096, 2048]
  const float* encw = (const float*)d_in[1];   // [2048, 2048]
  const float* encb = (const float*)d_in[2];   // [2048]
  const float* cb   = (const float*)d_in[3];   // [8, 8192, 256]
  const float* decw = (const float*)d_in[4];   // [2048, 2048]
  const float* decb = (const float*)d_in[5];   // [2048]

  float* out = (float*)d_out;
  const size_t OUT_XR = 0;                       // x_recon  [4096*2048]
  const size_t OUT_Q  = (size_t)Bsz * SD;        // quantized [4096*2048]
  const size_t OUT_C  = OUT_Q + (size_t)Bsz * SD;// codes     [4096*8]
  const size_t OUT_P  = OUT_C + (size_t)Bsz * SSPL; // perplexity [1]

  // workspace layout (bytes)
  char* ws = (char*)d_ws;
  unsigned short* x_bf  = (unsigned short*)(ws + 0);                 // 16 MB
  unsigned short* ew_bf = (unsigned short*)(ws + (16u << 20));       //  8 MB
  unsigned short* dw_bf = (unsigned short*)(ws + (24u << 20));       //  8 MB
  unsigned short* cb_bf = (unsigned short*)(ws + (32u << 20));       // 32 MB
  unsigned short* z_bf  = (unsigned short*)(ws + (64u << 20));       // 16 MB
  unsigned short* q_bf  = (unsigned short*)(ws + (80u << 20));       // 16 MB
  float*          esq   = (float*)         (ws + (96u << 20));       // 256 KB
  int*            codes = (int*)           (ws + (96u << 20) + (1u << 18)); // 128 KB
  int*            cnts  = (int*)           (ws + (96u << 20) + (1u << 18) + (1u << 17)); // 256 KB

  // 1. conversions
  cvt_f32_bf16_kernel<<<(Bsz * INDIM / 4 + 255) / 256, 256, 0, stream>>>(x, x_bf, Bsz * INDIM / 4);
  cvt_f32_bf16_kernel<<<(SD * INDIM / 4 + 255) / 256, 256, 0, stream>>>(encw, ew_bf, SD * INDIM / 4);
  cvt_f32_bf16_kernel<<<(INDIM * SD / 4 + 255) / 256, 256, 0, stream>>>(decw, dw_bf, INDIM * SD / 4);
  cvt_f32_bf16_kernel<<<(SSPL * KCODE * DDIM / 4 + 255) / 256, 256, 0, stream>>>(cb, cb_bf, SSPL * KCODE * DDIM / 4);

  // 2. e_sq
  esq_kernel<<<SSPL * KCODE / 8, 256, 0, stream>>>(cb, esq);

  // 3. encoder GEMM -> z (bf16)
  gemm_nt_bias_kernel<<<dim3(SD / 64, Bsz / 128), 256, 0, stream>>>(
      x_bf, ew_bf, encb, (void*)z_bf, INDIM, SD, 1);

  // 4. fused distances + argmin -> codes
  vq_dist_argmin_kernel<<<(SSPL * (Bsz / 16)) / 8, 256, 0, stream>>>(
      z_bf, esq, cb_bf, codes);

  // 5. gather quantized (f32 out + bf16) and codes (float out)
  vq_gather_kernel<<<(Bsz * SSPL) / 8, 256, 0, stream>>>(
      codes, cb, out + OUT_Q, q_bf, out + OUT_C);

  // 6. decoder GEMM -> x_recon (f32 out)
  gemm_nt_bias_kernel<<<dim3(INDIM / 64, Bsz / 128), 256, 0, stream>>>(
      q_bf, dw_bf, decb, (void*)(out + OUT_XR), SD, INDIM, 0);

  // 7. histogram + perplexity
  zero_i32_kernel<<<(SSPL * KCODE + 255) / 256, 256, 0, stream>>>(cnts, SSPL * KCODE);
  hist_kernel<<<(Bsz * SSPL + 255) / 256, 256, 0, stream>>>(codes, cnts);
  ppl_kernel<<<1, 256, 0, stream>>>(cnts, out + OUT_P);
}
